// MLALayer_17609365914526
// MI455X (gfx1250) — matrix operand order repaired and checked by reading, not compiled, not run on hardware
//
#include <hip/hip_runtime.h>
#include <hip/hip_bf16.h>

// ---------------- problem constants (from reference) ----------------
constexpr int kB   = 8;       // batch
constexpr int kT   = 4096;    // seq len
constexpr int kDM  = 512;     // d_model
constexpr int kNH  = 8;       // heads
constexpr int kHD  = 64;      // head dim
constexpr int kRK  = 128;     // kv rank
constexpr int kBT  = kB * kT; // 32768 rows for dense GEMMs

// epilogue modes for the generic GEMM
constexpr int MODE_F32     = 0;  // C0 = float*  (M,N) + bias
constexpr int MODE_BF16    = 1;  // C0 = bf16*   (M,N) + bias
constexpr int MODE_ROPE_Q  = 2;  // C0 = q  (B,H,T,64) bf16, rope+elu+1
constexpr int MODE_ROPE_KV = 3;  // C0 = kT (B,H,64,T), C1 = vT (B,H,64,T) bf16

// ---------------- types ----------------
typedef __bf16 bf16_t;
typedef __attribute__((ext_vector_type(16))) __bf16 v16bf;
typedef __attribute__((ext_vector_type(8)))  __bf16 v8bf;
typedef __attribute__((ext_vector_type(8)))  float  v8f;

static __device__ inline float bf2f(bf16_t v) {
    unsigned short u = __builtin_bit_cast(unsigned short, v);
    unsigned int   w = ((unsigned int)u) << 16;
    return __builtin_bit_cast(float, w);
}
static __device__ inline bf16_t f2bf(float f) {
    unsigned int u = __builtin_bit_cast(unsigned int, f);
    u += 0x7FFFu + ((u >> 16) & 1u);               // round to nearest even
    unsigned short s = (unsigned short)(u >> 16);
    return __builtin_bit_cast(bf16_t, s);
}

// CDNA5 async copy: global -> LDS, 16 bytes per lane, tracked by ASYNCcnt.
static __device__ inline void async_copy_b128(const bf16_t* lds_dst, const bf16_t* g) {
    unsigned int       l = (unsigned int)(uintptr_t)lds_dst;       // LDS byte offset
    unsigned long long a = (unsigned long long)(uintptr_t)g;       // global VA
    asm volatile("global_load_async_to_lds_b128 %0, %1, off"
                 :: "v"(l), "v"(a) : "memory");
}
static __device__ inline void wait_async0() {
    asm volatile("s_wait_asynccnt 0x0" ::: "memory");
}

// A fragment (16x32 bf16, M x K): lane holds row (lane&15), contiguous 16
// K-values starting at K + 16*(lane>=16).
static __device__ inline v16bf load_afrag(const bf16_t* p) {
    v8bf lo = *(const v8bf*)(p);
    v8bf hi = *(const v8bf*)(p + 8);
    v16bf r;
#pragma unroll
    for (int i = 0; i < 8; ++i) { r[i] = lo[i]; r[i + 8] = hi[i]; }
    return r;
}
// B fragment: lane holds column (lane&15), contiguous 16 K-values starting
// at K + 16*(lane>=16).
static __device__ inline v16bf load_bfrag(const bf16_t* p) {
    v8bf lo = *(const v8bf*)(p);
    v8bf hi = *(const v8bf*)(p + 8);
    v16bf r;
#pragma unroll
    for (int i = 0; i < 8; ++i) { r[i] = lo[i]; r[i + 8] = hi[i]; }
    return r;
}

// RoPE rotation + elu+1 on one (d, d+32) pair, hardware transcendentals.
static __device__ inline void rope_pair(float x1, float x2, int t, int i,
                                        float& o1, float& o2) {
    // 1/10000^(i/32) = exp2(-i * log2(10000)/32)
    const float inv_freq = exp2f(-(float)i * (13.2877123795f / 32.0f));
    float s, c;
    __sincosf((float)t * inv_freq, &s, &c);
    o1 = x1 * c - x2 * s;
    o2 = x1 * s + x2 * c;
    o1 = (o1 > 0.0f) ? (o1 + 1.0f) : __expf(o1);   // elu(x)+1
    o2 = (o2 > 0.0f) ? (o2 + 1.0f) : __expf(o2);
}

// ---------------- generic batched WMMA GEMM, LDS-staged B ----------------
// C = A[M,K] (bf16 row-major) * Bt[N,K]^T (bf16, pre-transposed) + bias.
// block = 128 threads (4 waves); wave tile 16x64 (4 accumulators); block tile
// 64x64. Pipeline stage = 64 K: 64x64 B tile (8KB) staged in LDS by async
// copies, double-buffered -> 8 WMMAs per barrier pair, 4x less global B
// traffic. grid.x = (M/64)*(N/64), grid.y = batch. K % 64 == 0 required.
constexpr int LDSP = 72;   // padded row (bf16): 144B rows -> conflict-free ds_load_b128
__global__ void __launch_bounds__(128)
wmma_gemm_bf16(const bf16_t* __restrict__ A, const bf16_t* __restrict__ Bt,
               const float* __restrict__ bias,
               void* __restrict__ C0, void* __restrict__ C1,
               int M, int N, int K, int mode,
               long long sA, long long sB, long long sC)
{
    __shared__ __align__(64) bf16_t ldsB[2][64][LDSP];

    const int lane = threadIdx.x & 31;
    const int wave = threadIdx.x >> 5;
    const int tid  = threadIdx.x;
    const int batch = blockIdx.y;
    A  += (size_t)batch * (size_t)sA;
    Bt += (size_t)batch * (size_t)sB;
    const size_t cOff = (size_t)batch * (size_t)sC;

    const int nBlocks = N >> 6;
    const int bm = blockIdx.x / nBlocks;
    const int bn = blockIdx.x - bm * nBlocks;
    const int mBase = (bm << 6) + (wave << 4);
    const int nBase = bn << 6;

    const int l15 = lane & 15;
    const int hi  = lane >> 4;

    const bf16_t* ap = A + (size_t)(mBase + l15) * K + hi * 16;

    // staging: 64 rows x 64 bf16 = 8KB = 512 x 16B chunks; 4 per thread.
    // chunk ci: row n = ci>>3, 16B part p = ci&7.
    int cn[4], cp[4];
    const bf16_t* cg[4];
    bf16_t* cl0[4];
    bf16_t* cl1[4];
#pragma unroll
    for (int j = 0; j < 4; ++j) {
        const int ci = tid + j * 128;
        cn[j] = ci >> 3;
        cp[j] = ci & 7;
        cg[j]  = Bt + (size_t)(nBase + cn[j]) * K + cp[j] * 8;
        cl0[j] = &ldsB[0][cn[j]][cp[j] * 8];
        cl1[j] = &ldsB[1][cn[j]][cp[j] * 8];
    }

    // prologue: stage k=0 into buffer 0
#pragma unroll
    for (int j = 0; j < 4; ++j) async_copy_b128(cl0[j], cg[j]);

    v8f acc0 = {}, acc1 = {}, acc2 = {}, acc3 = {};
    int buf = 0;
    for (int k = 0; k < K; k += 64) {
        wait_async0();          // this wave's loads for `buf` are done
        __syncthreads();        // every wave's loads for `buf` are done
        if (k + 64 < K) {       // prefetch next stage, overlapped with WMMAs
#pragma unroll
            for (int j = 0; j < 4; ++j)
                async_copy_b128(buf ? cl0[j] : cl1[j], cg[j] + k + 64);
        }
#pragma unroll
        for (int kk = 0; kk < 64; kk += 32) {
            v16bf a  = load_afrag(ap + k + kk);
            v16bf b0 = load_bfrag(&ldsB[buf][l15     ][kk + hi * 16]);
            v16bf b1 = load_bfrag(&ldsB[buf][l15 + 16][kk + hi * 16]);
            v16bf b2 = load_bfrag(&ldsB[buf][l15 + 32][kk + hi * 16]);
            v16bf b3 = load_bfrag(&ldsB[buf][l15 + 48][kk + hi * 16]);
            acc0 = __builtin_amdgcn_wmma_f32_16x16x32_bf16(false, a, false, b0, (short)0, acc0, false, false);
            acc1 = __builtin_amdgcn_wmma_f32_16x16x32_bf16(false, a, false, b1, (short)0, acc1, false, false);
            acc2 = __builtin_amdgcn_wmma_f32_16x16x32_bf16(false, a, false, b2, (short)0, acc2, false, false);
            acc3 = __builtin_amdgcn_wmma_f32_16x16x32_bf16(false, a, false, b3, (short)0, acc3, false, false);
        }
        __syncthreads();        // reads of `buf` done before it is overwritten
        buf ^= 1;
    }

    // ---------------- epilogue ----------------
    const int rBase = mBase + (hi << 3);  // C/D: VGPR i -> M = i + 8*(lane>=16)
    const int c0 = nBase + l15;
    float bb0 = 0.f, bb1 = 0.f, bb2 = 0.f, bb3 = 0.f;
    if (bias) { bb0 = bias[c0]; bb1 = bias[c0 + 16]; bb2 = bias[c0 + 32]; bb3 = bias[c0 + 48]; }

    if (mode == MODE_F32) {
        float* Cf = (float*)C0;
#pragma unroll
        for (int i = 0; i < 8; ++i) {
            const size_t o = cOff + (size_t)(rBase + i) * N + c0;
            Cf[o] = acc0[i] + bb0; Cf[o + 16] = acc1[i] + bb1;
            Cf[o + 32] = acc2[i] + bb2; Cf[o + 48] = acc3[i] + bb3;
        }
    } else if (mode == MODE_BF16) {
        bf16_t* Cb = (bf16_t*)C0;
#pragma unroll
        for (int i = 0; i < 8; ++i) {
            const size_t o = cOff + (size_t)(rBase + i) * N + c0;
            Cb[o] = f2bf(acc0[i] + bb0); Cb[o + 16] = f2bf(acc1[i] + bb1);
            Cb[o + 32] = f2bf(acc2[i] + bb2); Cb[o + 48] = f2bf(acc3[i] + bb3);
        }
    } else if (mode == MODE_ROPE_Q) {
        // cols nBase..nBase+63 = one head h; dims (l15, +16, +32, +48).
        // rope pairs (d, d+32), freq index d -> (acc0,acc2)@l15, (acc1,acc3)@l15+16
        bf16_t* qd = (bf16_t*)C0;
        const int h = (nBase >> 6) & (kNH - 1);
#pragma unroll
        for (int i = 0; i < 8; ++i) {
            const int r = rBase + i;
            const int t = r & (kT - 1), b = r >> 12;
            float o1, o2, o3, o4;
            rope_pair(acc0[i] + bb0, acc2[i] + bb2, t, l15,      o1, o2);
            rope_pair(acc1[i] + bb1, acc3[i] + bb3, t, l15 + 16, o3, o4);
            bf16_t* dst = qd + ((size_t)(b * kNH + h) * kT + t) * kHD;
            dst[l15]      = f2bf(o1); dst[l15 + 32] = f2bf(o2);
            dst[l15 + 16] = f2bf(o3); dst[l15 + 48] = f2bf(o4);
        }
    } else {  // MODE_ROPE_KV: per head 128 cols = [k(64) | v(64)]
        bf16_t* kd = (bf16_t*)C0;
        bf16_t* vd = (bf16_t*)C1;
        const int h = nBase >> 7;
        const bool isK = ((nBase >> 6) & 1) == 0;
#pragma unroll
        for (int i = 0; i < 8; ++i) {
            const int r = rBase + i;
            const int t = r & (kT - 1), b = r >> 12;
            const size_t base = (size_t)(b * kNH + h) * kHD;
            if (isK) {
                float o1, o2, o3, o4;
                rope_pair(acc0[i] + bb0, acc2[i] + bb2, t, l15,      o1, o2);
                rope_pair(acc1[i] + bb1, acc3[i] + bb3, t, l15 + 16, o3, o4);
                kd[(base + l15)      * kT + t] = f2bf(o1);
                kd[(base + l15 + 32) * kT + t] = f2bf(o2);
                kd[(base + l15 + 16) * kT + t] = f2bf(o3);
                kd[(base + l15 + 48) * kT + t] = f2bf(o4);
            } else {
                vd[(base + l15)      * kT + t] = f2bf(acc0[i] + bb0);
                vd[(base + l15 + 16) * kT + t] = f2bf(acc1[i] + bb1);
                vd[(base + l15 + 32) * kT + t] = f2bf(acc2[i] + bb2);
                vd[(base + l15 + 48) * kT + t] = f2bf(acc3[i] + bb3);
            }
        }
    }
}

// ---------------- attention out: out = q @ ctxT^T, scaled by z ----------------
// per batch bh: A = q (T x 64), Bt = ctxT (64 x 64, tiny -> direct loads, L2
// resident). epilogue scatters to attn (b,t,h,d) bf16 scaled by z[t].
// grid = (T/64, B*H), K = 64 fully unrolled.
__global__ void __launch_bounds__(128)
attn_out_kernel(const bf16_t* __restrict__ qbf, const bf16_t* __restrict__ ctxT,
                const float* __restrict__ zbuf, bf16_t* __restrict__ attn)
{
    const int lane = threadIdx.x & 31;
    const int wave = threadIdx.x >> 5;
    const int bh = blockIdx.y;
    const int b = bh >> 3, h = bh & 7;
    const bf16_t* A  = qbf  + (size_t)bh * kT * kHD;
    const bf16_t* Bt = ctxT + (size_t)bh * kHD * kHD;
    const float*  z  = zbuf + (size_t)bh * kT;

    const int mBase = (blockIdx.x << 6) + (wave << 4);
    const int l15 = lane & 15;
    const int hi  = lane >> 4;

    const bf16_t* ap = A + (size_t)(mBase + l15) * kHD + hi * 16;
    v8f acc0 = {}, acc1 = {}, acc2 = {}, acc3 = {};
#pragma unroll
    for (int k = 0; k < kHD; k += 32) {
        v16bf a  = load_afrag(ap + k);
        v16bf b0 = load_bfrag(Bt + (size_t)(l15)      * kHD + hi * 16 + k);
        v16bf b1 = load_bfrag(Bt + (size_t)(16 + l15) * kHD + hi * 16 + k);
        v16bf b2 = load_bfrag(Bt + (size_t)(32 + l15) * kHD + hi * 16 + k);
        v16bf b3 = load_bfrag(Bt + (size_t)(48 + l15) * kHD + hi * 16 + k);
        acc0 = __builtin_amdgcn_wmma_f32_16x16x32_bf16(false, a, false, b0, (short)0, acc0, false, false);
        acc1 = __builtin_amdgcn_wmma_f32_16x16x32_bf16(false, a, false, b1, (short)0, acc1, false, false);
        acc2 = __builtin_amdgcn_wmma_f32_16x16x32_bf16(false, a, false, b2, (short)0, acc2, false, false);
        acc3 = __builtin_amdgcn_wmma_f32_16x16x32_bf16(false, a, false, b3, (short)0, acc3, false, false);
    }

    const int rBase = mBase + (hi << 3);
#pragma unroll
    for (int i = 0; i < 8; ++i) {
        const int t = rBase + i;
        const float zz = z[t];
        bf16_t* dst = attn + ((size_t)(b * kT + t) * kNH + h) * kHD + l15;
        dst[0]  = f2bf(acc0[i] * zz);
        dst[16] = f2bf(acc1[i] * zz);
        dst[32] = f2bf(acc2[i] * zz);
        dst[48] = f2bf(acc3[i] * zz);
    }
}

// ---------------- elementwise / conversion kernels ----------------
__global__ void cvt_f32_to_bf16(const float* __restrict__ in, bf16_t* __restrict__ out, int n) {
    int i = blockIdx.x * blockDim.x + threadIdx.x;
    if (i < n) out[i] = f2bf(in[i]);
}

// W (K,N) row-major fp32 -> Wt (N,K) row-major bf16
__global__ void cvt_transpose(const float* __restrict__ W, bf16_t* __restrict__ Wt, int K, int N) {
    int i = blockIdx.x * blockDim.x + threadIdx.x;
    if (i < K * N) {
        int k = i / N, n = i - k * N;
        Wt[(size_t)n * K + k] = f2bf(W[i]);
    }
}

// k_sum[bh,d] = sum_t kT[bh,d,t]
__global__ void ksum_kernel(const bf16_t* __restrict__ kTr, float* __restrict__ ksum) {
    int idx = blockIdx.x * blockDim.x + threadIdx.x;
    if (idx < kB * kNH * kHD) {
        const bf16_t* row = kTr + (size_t)idx * kT;
        float s = 0.0f;
        for (int t = 0; t < kT; t += 8) {
            v8bf v = *(const v8bf*)(row + t);
#pragma unroll
            for (int j = 0; j < 8; ++j) s += bf2f(v[j]);
        }
        ksum[idx] = s;
    }
}

// z[bh,t] = 1 / (q[bh,t,:] . ksum[bh,:] + 1e-6)
__global__ void z_kernel(const bf16_t* __restrict__ qbf, const float* __restrict__ ksum,
                         float* __restrict__ zbuf) {
    int idx = blockIdx.x * blockDim.x + threadIdx.x;  // bh*T + t
    int bh = idx >> 12;
    int t  = idx & (kT - 1);
    const bf16_t* qr = qbf + ((size_t)bh * kT + t) * kHD;
    const float* ks = ksum + (size_t)bh * kHD;
    float d = 0.0f;
#pragma unroll 8
    for (int j = 0; j < kHD; ++j) d += bf2f(qr[j]) * ks[j];
    zbuf[idx] = 1.0f / (d + 1e-6f);
}

// ---------------- host launcher ----------------
extern "C" void kernel_launch(void* const* d_in, const int* in_sizes, int n_in,
                              void* d_out, int out_size, void* d_ws, size_t ws_size,
                              hipStream_t stream) {
    const float* x  = (const float*)d_in[0];
    const float* Wq = (const float*)d_in[1];
    const float* bq = (const float*)d_in[2];
    const float* Wd = (const float*)d_in[3];
    const float* bd = (const float*)d_in[4];
    const float* Wu = (const float*)d_in[5];
    const float* bu = (const float*)d_in[6];
    const float* Wo = (const float*)d_in[7];
    const float* bo = (const float*)d_in[8];
    float* out = (float*)d_out;

    unsigned char* wp = (unsigned char*)d_ws;
    auto carve = [&](size_t bytes) -> void* {
        void* r = (void*)wp;
        wp += (bytes + 255) & ~(size_t)255;
        return r;
    };
    bf16_t* xbf  = (bf16_t*)carve((size_t)kBT * kDM * 2);
    bf16_t* WqT  = (bf16_t*)carve((size_t)kDM * kDM * 2);
    bf16_t* WdT  = (bf16_t*)carve((size_t)kRK * kDM * 2);
    bf16_t* WuT  = (bf16_t*)carve((size_t)(2 * kDM) * kRK * 2);
    bf16_t* WoT  = (bf16_t*)carve((size_t)kDM * kDM * 2);
    bf16_t* c_bf = (bf16_t*)carve((size_t)kBT * kRK * 2);
    bf16_t* qbf  = (bf16_t*)carve((size_t)kBT * kDM * 2);
    bf16_t* kTr  = (bf16_t*)carve((size_t)kBT * kDM * 2);
    bf16_t* vTr  = (bf16_t*)carve((size_t)kBT * kDM * 2);
    bf16_t* ctxT = (bf16_t*)carve((size_t)kB * kNH * kHD * kHD * 2);
    float*  ksum = (float*) carve((size_t)kB * kNH * kHD * 4);
    float*  zbuf = (float*) carve((size_t)kB * kNH * kT * 4);
    bf16_t* attn = (bf16_t*)carve((size_t)kBT * kDM * 2);

    const int EB = 256;
    // 1) precision conversion / weight transposes
    cvt_f32_to_bf16<<<(kBT * kDM + EB - 1) / EB, EB, 0, stream>>>(x, xbf, kBT * kDM);
    cvt_transpose<<<(kDM * kDM + EB - 1) / EB, EB, 0, stream>>>(Wq, WqT, kDM, kDM);
    cvt_transpose<<<(kDM * kRK + EB - 1) / EB, EB, 0, stream>>>(Wd, WdT, kDM, kRK);
    cvt_transpose<<<(kRK * 2 * kDM + EB - 1) / EB, EB, 0, stream>>>(Wu, WuT, kRK, 2 * kDM);
    cvt_transpose<<<(kDM * kDM + EB - 1) / EB, EB, 0, stream>>>(Wo, WoT, kDM, kDM);

    // 2) q projection with fused RoPE + elu+1 -> q (B,H,T,64) bf16
    wmma_gemm_bf16<<<dim3((kBT / 64) * (kDM / 64), 1), 128, 0, stream>>>(
        xbf, WqT, bq, qbf, nullptr, kBT, kDM, kDM, MODE_ROPE_Q, 0, 0, 0);
    // 3) down-projection c = x Wd + bd -> bf16
    wmma_gemm_bf16<<<dim3((kBT / 64) * (kRK / 64), 1), 128, 0, stream>>>(
        xbf, WdT, bd, c_bf, nullptr, kBT, kRK, kDM, MODE_BF16, 0, 0, 0);
    // 4) up-projection with fused RoPE(k)+elu+1 and k/v transpose
    wmma_gemm_bf16<<<dim3((kBT / 64) * (2 * kDM / 64), 1), 128, 0, stream>>>(
        c_bf, WuT, bu, kTr, vTr, kBT, 2 * kDM, kRK, MODE_ROPE_KV, 0, 0, 0);

    // 5) k_sum and z
    ksum_kernel<<<(kB * kNH * kHD + EB - 1) / EB, EB, 0, stream>>>(kTr, ksum);
    z_kernel<<<(kB * kNH * kT) / EB, EB, 0, stream>>>(qbf, ksum, zbuf);

    // 6) ctxT[bh] = vT (64,T) x kT(64,T)^T : batched over B*H, K = T = 4096
    wmma_gemm_bf16<<<dim3(1, kB * kNH), 128, 0, stream>>>(
        vTr, kTr, nullptr, ctxT, nullptr, kHD, kHD, kT, MODE_BF16,
        (long long)kHD * kT, (long long)kHD * kT, (long long)kHD * kHD);

    // 7) out = (q @ ctx) * z -> attn (B,T,H,64) bf16
    attn_out_kernel<<<dim3(kT / 64, kB * kNH), 128, 0, stream>>>(qbf, ctxT, zbuf, attn);

    // 8) final projection -> fp32 d_out
    wmma_gemm_bf16<<<dim3((kBT / 64) * (kDM / 64), 1), 128, 0, stream>>>(
        attn, WoT, bo, out, nullptr, kBT, kDM, kDM, MODE_F32, 0, 0, 0);
}
